// IEViT_34308198760654
// MI455X (gfx1250) — matrix operand order repaired and checked by reading, not compiled, not run on hardware
//
#include <hip/hip_runtime.h>
#include <hip/hip_bf16.h>

// ============================================================================
// IEViT adaptive patch embed for MI455X (gfx1250).
//
//  x:(32,3,512,512) f32, patch_weight:(768,3,32,32) f32, bias:(768) f32
//  out:(32, 640, 768) f32   (static token structure: even patches subdivided)
//
//  1) x -> bf16, patch_weight -> W1 [e][k] bf16, averaged kernel -> W2 bf16
//  2) GEMM-full : odd patches only   (M=4096 , K=3072, N=768) -> scatter out
//  3) GEMM-sub  : even-patch quads   (M=16384, K=768 , N=768) -> scatter out
//  v_wmma_f32_16x16x32_bf16, f32 accumulate. Double-buffered LDS tiles fed by
//  GLOBAL_LOAD_ASYNC_TO_LDS_B128 (ASYNCcnt) when available, bank-conflict-free
//  80B row pitch, nontemporal result stores, global_prefetch of tile kb+2.
// ============================================================================

typedef __attribute__((ext_vector_type(16))) __bf16 v16bf;
typedef __attribute__((ext_vector_type(8)))  float  v8f;

union FragV { uint4 u[2]; v16bf v; };

#if defined(__has_builtin)
#  if __has_builtin(__builtin_amdgcn_global_load_async_to_lds_b128) && \
      __has_builtin(__builtin_amdgcn_s_wait_asynccnt)
#    define USE_ASYNC 1
#  endif
#endif
#ifndef USE_ASYNC
#  define USE_ASYNC 0
#endif

#if USE_ASYNC
typedef __attribute__((__vector_size__(16))) int v4i;
typedef __attribute__((address_space(1))) v4i glb_v4i;   // printed as "__device__"
typedef __attribute__((address_space(3))) v4i lds_v4i;   // printed as "__shared__"
__device__ __forceinline__ void async_cp16(const unsigned short* g, unsigned short* l) {
  __builtin_amdgcn_global_load_async_to_lds_b128((glb_v4i*)(void*)g,
                                                 (lds_v4i*)(void*)l, 0, 0);
}
#endif

__device__ __forceinline__ unsigned short f2bf(float f) {
  unsigned int u = __float_as_uint(f);
  u += 0x7FFFu + ((u >> 16) & 1u);          // round-to-nearest-even
  return (unsigned short)(u >> 16);
}

// ---------------------------------------------------------------- converters
__global__ __launch_bounds__(256)
void k_cvt_bf16(const float* __restrict__ src, unsigned short* __restrict__ dst,
                int n4) {
  int t = blockIdx.x * 256 + threadIdx.x;
  if (t >= n4) return;
  float4 f = ((const float4*)src)[t];
  ushort4 o;
  o.x = f2bf(f.x); o.y = f2bf(f.y); o.z = f2bf(f.z); o.w = f2bf(f.w);
  ((ushort4*)dst)[t] = o;
}

// w_sub[e,c,ii,jj] = mean of 2x2 quadrants -> W2[e][c*256+ii*16+jj]
__global__ __launch_bounds__(256)
void k_build_w2(const float* __restrict__ pw, unsigned short* __restrict__ w2) {
  int t = blockIdx.x * 256 + threadIdx.x;       // 768*768 threads
  if (t >= 768 * 768) return;
  int e = t / 768, k2 = t - e * 768;
  int c = k2 >> 8, rem = k2 & 255, ii = rem >> 4, jj = rem & 15;
  const float* base = pw + ((size_t)(e * 3 + c) << 10);
  float s = base[ii * 32 + jj]          + base[ii * 32 + jj + 16]
          + base[(ii + 16) * 32 + jj]   + base[(ii + 16) * 32 + jj + 16];
  w2[t] = f2bf(0.25f * s);
}

// ---------------------------------------------------------------- GEMM common
#define BM   128
#define BN   128
#define LDK  40          // padded LDS row pitch (bf16): 80B -> conflict-free

// copy one 32B half-row into LDS tile (async if available)
#if USE_ASYNC
#define TILE_CP32(dst, src) do { async_cp16((src), (dst)); \
                                 async_cp16((src) + 8, (dst) + 8); } while (0)
#else
#define TILE_CP32(dst, src) do { uint4 _t0 = ((const uint4*)(src))[0]; \
                                 uint4 _t1 = ((const uint4*)(src))[1]; \
                                 ((uint4*)(dst))[0] = _t0;             \
                                 ((uint4*)(dst))[1] = _t1; } while (0)
#endif

// ------------------------- GEMM 1: full 32x32 patches (odd patches only) ----
// M = 32*128 = 4096, K = 3072, N = 768.  A row m=(b, odd patch), k=(c,i,j).
__global__ __launch_bounds__(256)
void k_gemm_full(const unsigned short* __restrict__ xb,
                 const unsigned short* __restrict__ w1,
                 const float* __restrict__ bias,
                 float* __restrict__ out) {
  __shared__ __align__(16) unsigned short a_sh[2][BM * LDK];
  __shared__ __align__(16) unsigned short b_sh[2][BN * LDK];

  const int tid  = threadIdx.x;
  const int lane = tid & 31, wid = tid >> 5;
  const int mw = wid >> 2, nw = wid & 3;          // 2 x 4 wave grid
  const int lhalf = lane >> 4, lmod = lane & 15;

  const int m0 = blockIdx.x * BM;
  const int n0 = blockIdx.y * BN;

  // --- tile loader coordinates: 2 threads per 64B row
  const int lrow = tid >> 1;          // 0..127
  const int lseg = tid & 1;           // which 32B half
  const int am  = m0 + lrow;
  const int ab  = am >> 7, apo = am & 127;
  const int api = 2 * apo + 1;                    // odd patch index
  const int ar  = api >> 4, aq = api & 15;
  const unsigned short* aRow0 =
      xb + (size_t)ab * 786432 + (size_t)ar * 16384 + aq * 32 + lseg * 16;
  const unsigned short* bRow0 =
      w1 + (size_t)(n0 + lrow) * 3072 + lseg * 16;

  // k = c*1024 + i*32 + j : c = kb>>5, i = kb&31, j contiguous
  auto aSrcAt = [&](int kb) {
    return aRow0 + ((size_t)(kb >> 5) << 18) + ((kb & 31) << 9);
  };
  auto issueTile = [&](int kb, int buf) {
    TILE_CP32(&a_sh[buf][lrow * LDK + lseg * 16], aSrcAt(kb));
    TILE_CP32(&b_sh[buf][lrow * LDK + lseg * 16], bRow0 + (kb << 5));
  };

  v8f acc[4][2] = {};
  issueTile(0, 0);

  for (int kb = 0; kb < 96; ++kb) {
    const int cur = kb & 1;
#if USE_ASYNC
    __builtin_amdgcn_s_wait_asynccnt(0);
#endif
    __syncthreads();
    if (kb + 1 < 96) issueTile(kb + 1, cur ^ 1);
    if (kb + 2 < 96) {
      __builtin_prefetch(aSrcAt(kb + 2), 0, 1);
      __builtin_prefetch(bRow0 + ((kb + 2) << 5), 0, 1);
    }

    const unsigned short* As = a_sh[cur];
    const unsigned short* Bs = b_sh[cur];
    FragV bf[2], af[4];
#pragma unroll
    for (int ni = 0; ni < 2; ++ni) {              // B: lane=col, 16 contig K
      const uint4* p =
          (const uint4*)&Bs[(nw * 32 + ni * 16 + lmod) * LDK + lhalf * 16];
      bf[ni].u[0] = p[0]; bf[ni].u[1] = p[1];
    }
#pragma unroll
    for (int mi = 0; mi < 4; ++mi) {              // A: lane=row, 16 contig K
      const int row = mw * 64 + mi * 16 + lmod;
      af[mi].u[0] = *(const uint4*)&As[row * LDK + lhalf * 16];
      af[mi].u[1] = *(const uint4*)&As[row * LDK + lhalf * 16 + 8];
    }
#pragma unroll
    for (int mi = 0; mi < 4; ++mi)
#pragma unroll
      for (int ni = 0; ni < 2; ++ni)
        acc[mi][ni] = __builtin_amdgcn_wmma_f32_16x16x32_bf16(
            false, af[mi].v, false, bf[ni].v, (short)0, acc[mi][ni],
            false, false);
    __syncthreads();
  }

  // epilogue: odd patch i -> token (5i+3)/2 ; nontemporal scatter with bias
#pragma unroll
  for (int ni = 0; ni < 2; ++ni) {
    const int e  = n0 + nw * 32 + ni * 16 + lmod;
    const float bv = bias[e];
#pragma unroll
    for (int mi = 0; mi < 4; ++mi) {
#pragma unroll
      for (int v = 0; v < 8; ++v) {
        int m  = m0 + mw * 64 + mi * 16 + 8 * lhalf + v;
        int b  = m >> 7, po = m & 127;
        int ip = 2 * po + 1;
        int t  = (5 * ip + 3) >> 1;
        __builtin_nontemporal_store(acc[mi][ni][v] + bv,
                                    &out[(size_t)(b * 640 + t) * 768 + e]);
      }
    }
  }
}

// ------------------- GEMM 2: 16x16 sub-patches (even-patch quadrants only) --
// M = 32*512 = 16384, K = 768, N = 768.  Rows: (b, rr, 16 allowed qq).
__global__ __launch_bounds__(256)
void k_gemm_sub(const unsigned short* __restrict__ xb,
                const unsigned short* __restrict__ w2,
                const float* __restrict__ bias,
                float* __restrict__ out) {
  __shared__ __align__(16) unsigned short a_sh[2][BM * LDK];
  __shared__ __align__(16) unsigned short b_sh[2][BN * LDK];

  const int tid  = threadIdx.x;
  const int lane = tid & 31, wid = tid >> 5;
  const int mw = wid >> 2, nw = wid & 3;
  const int lhalf = lane >> 4, lmod = lane & 15;

  const int m0 = blockIdx.x * BM;
  const int n0 = blockIdx.y * BN;

  const int lrow = tid >> 1;
  const int lseg = tid & 1;
  const int am  = m0 + lrow;
  const int ab  = am >> 9, as = am & 511;
  const int arr = as >> 4, aqi = as & 15;
  const int aqq = ((aqi >> 1) << 2) + (aqi & 1);  // qq with qq//2 even
  const unsigned short* aRow0 =
      xb + (size_t)ab * 786432 + (size_t)arr * 8192 + aqq * 16;
  const unsigned short* bRow0 =
      w2 + (size_t)(n0 + lrow) * 768 + lseg * 16;

  // k = c*256 + i*16 + j : c = kb>>3 ; i = (kb&7)*2 + lseg ; j = 0..15
  auto aSrcAt = [&](int kb) {
    return aRow0 + ((size_t)(kb >> 3) << 18) + ((((kb & 7) << 1) + lseg) << 9);
  };
  auto issueTile = [&](int kb, int buf) {
    TILE_CP32(&a_sh[buf][lrow * LDK + lseg * 16], aSrcAt(kb));
    TILE_CP32(&b_sh[buf][lrow * LDK + lseg * 16], bRow0 + (kb << 5));
  };

  v8f acc[4][2] = {};
  issueTile(0, 0);

  for (int kb = 0; kb < 24; ++kb) {
    const int cur = kb & 1;
#if USE_ASYNC
    __builtin_amdgcn_s_wait_asynccnt(0);
#endif
    __syncthreads();
    if (kb + 1 < 24) issueTile(kb + 1, cur ^ 1);
    if (kb + 2 < 24) {
      __builtin_prefetch(aSrcAt(kb + 2), 0, 1);
      __builtin_prefetch(bRow0 + ((kb + 2) << 5), 0, 1);
    }

    const unsigned short* As = a_sh[cur];
    const unsigned short* Bs = b_sh[cur];
    FragV bf[2], af[4];
#pragma unroll
    for (int ni = 0; ni < 2; ++ni) {
      const uint4* p =
          (const uint4*)&Bs[(nw * 32 + ni * 16 + lmod) * LDK + lhalf * 16];
      bf[ni].u[0] = p[0]; bf[ni].u[1] = p[1];
    }
#pragma unroll
    for (int mi = 0; mi < 4; ++mi) {
      const int row = mw * 64 + mi * 16 + lmod;
      af[mi].u[0] = *(const uint4*)&As[row * LDK + lhalf * 16];
      af[mi].u[1] = *(const uint4*)&As[row * LDK + lhalf * 16 + 8];
    }
#pragma unroll
    for (int mi = 0; mi < 4; ++mi)
#pragma unroll
      for (int ni = 0; ni < 2; ++ni)
        acc[mi][ni] = __builtin_amdgcn_wmma_f32_16x16x32_bf16(
            false, af[mi].v, false, bf[ni].v, (short)0, acc[mi][ni],
            false, false);
    __syncthreads();
  }

  // epilogue: even patch i = (rr/2)*16 + qq/2 ; token 5i/2 + 2*(rr&1) + (qq&1)
#pragma unroll
  for (int ni = 0; ni < 2; ++ni) {
    const int e  = n0 + nw * 32 + ni * 16 + lmod;
    const float bv = bias[e];
#pragma unroll
    for (int mi = 0; mi < 4; ++mi) {
#pragma unroll
      for (int v = 0; v < 8; ++v) {
        int m  = m0 + mw * 64 + mi * 16 + 8 * lhalf + v;
        int b  = m >> 9, s = m & 511;
        int rr = s >> 4, qi = s & 15;
        int ip = (rr >> 1) * 16 + ((qi >> 1) << 1);
        int t  = ((5 * ip) >> 1) + ((rr & 1) << 1) + (qi & 1);
        __builtin_nontemporal_store(acc[mi][ni][v] + bv,
                                    &out[(size_t)(b * 640 + t) * 768 + e]);
      }
    }
  }
}

// ---------------------------------------------------------------- launcher
extern "C" void kernel_launch(void* const* d_in, const int* in_sizes, int n_in,
                              void* d_out, int out_size, void* d_ws, size_t ws_size,
                              hipStream_t stream) {
  (void)in_sizes; (void)n_in; (void)out_size; (void)ws_size;

  const float* x  = (const float*)d_in[0];   // (32,3,512,512)
  const float* pw = (const float*)d_in[2];   // (768,3,32,32)
  const float* pb = (const float*)d_in[3];   // (768)
  float* out = (float*)d_out;                // (32,640,768)

  // workspace layout (bf16 elements)
  unsigned short* xb = (unsigned short*)d_ws;        // 25,165,824
  unsigned short* w1 = xb + 25165824;                //  2,359,296  [e][k=3072]
  unsigned short* w2 = w1 + 2359296;                 //    589,824  [e][k=768]

  k_cvt_bf16<<<(25165824 / 4 + 255) / 256, 256, 0, stream>>>(x, xb, 25165824 / 4);
  k_cvt_bf16<<<(2359296 / 4 + 255) / 256, 256, 0, stream>>>(pw, w1, 2359296 / 4);
  k_build_w2<<<(768 * 768 + 255) / 256, 256, 0, stream>>>(pw, w2);
  k_gemm_full<<<dim3(4096 / BM, 768 / BN), 256, 0, stream>>>(xb, w1, pb, out);
  k_gemm_sub<<<dim3(16384 / BM, 768 / BN), 256, 0, stream>>>(xb, w2, pb, out);
}
